// SAGENet_37778532336373
// MI455X (gfx1250) — matrix operand order repaired and checked by reading, not compiled, not run on hardware
//
#include <hip/hip_runtime.h>
#include <math.h>

// ---------------- problem constants (from reference) ----------------
#define N_NODES 50000
#define N_EDGES 600000
#define IN_CH   128
#define HID     256
#define OUT_F   256
#define OUT_CH  128

typedef __attribute__((ext_vector_type(16))) __bf16 v16bf;
typedef __attribute__((ext_vector_type(8)))  __bf16 v8bf;
typedef __attribute__((ext_vector_type(8)))  float  v8f;

// ---------------------------------------------------------------
// WMMA fragment builders for V_WMMA_F32_16X16X32_BF16 (wave32).
// A (16xK tile slice, MxK): lanes hold M = lane&15.
//   lane<16 -> K=k0+e, lane>=16 -> K=k0+16+e  (e = 0..15)
// B (Kx16 slice, from row-major bf16 weight W[n][k]): lanes hold N = lane&15,
//   lane<16 -> K=k0+e, lane>=16 -> K=k0+16+e  (e = 0..15)
// All inputs pre-converted to bf16 -> zero VALU cvt in the inner loop.
// ---------------------------------------------------------------
__device__ __forceinline__ v16bf load_a_frag(const __bf16* __restrict__ aRow,
                                             int k0, int khi) {
  const __bf16* p = aRow + k0 + khi * 16;
  v8bf lo = *(const v8bf*)(p);        // 16B LDS load
  v8bf hi = *(const v8bf*)(p + 8);    // 16B LDS load
  return __builtin_shufflevector(lo, hi, 0, 1, 2, 3, 4, 5, 6, 7,
                                 8, 9, 10, 11, 12, 13, 14, 15);
}

__device__ __forceinline__ v16bf load_b_frag(const __bf16* __restrict__ wRow,
                                             int k0, int khi) {
  return *(const v16bf*)(wRow + k0 + khi * 16);  // 32B global load
}

// acc += A(16xK tile, row m implied by lane) @ W[n][:K]^T   (bf16 in, f32 acc)
template <int KDIM>
__device__ __forceinline__ v8f wmma_rowtile(const __bf16* __restrict__ aRow,
                                            const __bf16* __restrict__ W,
                                            int n, int khi, v8f acc) {
  const __bf16* wRow = W + (size_t)n * KDIM;
#pragma unroll
  for (int k0 = 0; k0 < KDIM; k0 += 32) {
    v16bf a = load_a_frag(aRow, k0, khi);
    v16bf b = load_b_frag(wRow, k0, khi);
    acc = __builtin_amdgcn_wmma_f32_16x16x32_bf16(
        /*neg_a=*/false, a, /*neg_b=*/false, b,
        /*c_mod=*/(short)0, acc, /*reuse_a=*/false, /*reuse_b=*/false);
  }
  return acc;
}

// ---------------- zero workspace ----------------
__global__ void zero_kernel(float* __restrict__ p, long n) {
  long i = (long)blockIdx.x * blockDim.x + threadIdx.x;
  long stride = (long)gridDim.x * blockDim.x;
  for (; i < n; i += stride) p[i] = 0.0f;
}

// ---------------- fp32 -> bf16 weight pre-conversion (runs once per call) --
__global__ void cvt_bf16_kernel(const float* __restrict__ src,
                                __bf16* __restrict__ dst, int n) {
  int i = blockIdx.x * blockDim.x + threadIdx.x;
  if (i < n) dst[i] = (__bf16)src[i];
}

// ---------------- scatter kernels (segment mean numerators) ----------------
// conv1: all edges, 128 channels. 32 threads/edge, 4 ch each (float4 gather).
__global__ void scatter1_kernel(const float* __restrict__ x,
                                const int* __restrict__ src,
                                const int* __restrict__ dst,
                                float* __restrict__ agg,
                                float* __restrict__ cnt) {
  int idx = blockIdx.x * blockDim.x + threadIdx.x;
  int e = idx >> 5;
  if (e >= N_EDGES) return;
  int c = (idx & 31) * 4;
  int s = src[e], d = dst[e];
  float4 v = *(const float4*)&x[(size_t)s * IN_CH + c];
  float* ap = &agg[(size_t)d * IN_CH + c];
  atomicAdd(ap + 0, v.x);
  atomicAdd(ap + 1, v.y);
  atomicAdd(ap + 2, v.z);
  atomicAdd(ap + 3, v.w);
  if ((idx & 31) == 0) atomicAdd(&cnt[d], 1.0f);
}

// conv2: masked edges, 256 channels. 64 threads/edge, 4 ch each.
__global__ void scatter2_kernel(const float* __restrict__ h,
                                const int* __restrict__ src,
                                const int* __restrict__ dst,
                                const unsigned char* __restrict__ mask,
                                float* __restrict__ agg,
                                float* __restrict__ cnt) {
  int idx = blockIdx.x * blockDim.x + threadIdx.x;
  int e = idx >> 6;
  if (e >= N_EDGES) return;
  if (!mask[e]) return;
  int c = (idx & 63) * 4;
  int s = src[e], d = dst[e];
  float4 v = *(const float4*)&h[(size_t)s * HID + c];
  float* ap = &agg[(size_t)d * HID + c];
  atomicAdd(ap + 0, v.x);
  atomicAdd(ap + 1, v.y);
  atomicAdd(ap + 2, v.z);
  atomicAdd(ap + 3, v.w);
  if ((idx & 63) == 0) atomicAdd(&cnt[d], 1.0f);
}

// ---------------- fused layer 1 ----------------
// h1 = tanh( normalize(mean_agg @ W1l^T + b1 + x @ W1r^T) + x @ Ws^T + bs )
// Block: 16 nodes x 256 outputs. 256 threads = 8 waves, 2 n-tiles per wave.
__global__ __launch_bounds__(256) void layer1_kernel(
    const float* __restrict__ x, const float* __restrict__ agg,
    const float* __restrict__ cnt,
    const __bf16* __restrict__ W1l, const float* __restrict__ b1,
    const __bf16* __restrict__ W1r,
    const __bf16* __restrict__ Ws,  const float* __restrict__ bs,
    float* __restrict__ h1) {
  __shared__ __align__(16) __bf16 sA[16 * IN_CH];  // mean aggregated feats
  __shared__ __align__(16) __bf16 sX[16 * IN_CH];  // node feats
  __shared__ float rinv[16];
  __shared__ float rowsq[16];

  const int tid = threadIdx.x;
  const int node0 = blockIdx.x * 16;

  if (tid < 16) {
    rinv[tid]  = 1.0f / fmaxf(cnt[node0 + tid], 1.0f);
    rowsq[tid] = 0.0f;
  }
  __syncthreads();

  for (int i = tid; i < 16 * IN_CH; i += 256) {
    int r = i >> 7, k = i & (IN_CH - 1);
    sX[i] = (__bf16)x[(size_t)(node0 + r) * IN_CH + k];
    sA[i] = (__bf16)(agg[(size_t)(node0 + r) * IN_CH + k] * rinv[r]);
  }
  __syncthreads();

  const int wave = tid >> 5, lane = tid & 31;
  const int m = lane & 15, khi = lane >> 4;
  const __bf16* aRowA = sA + m * IN_CH;
  const __bf16* aRowX = sX + m * IN_CH;

  v8f accA[2], accB[2];
#pragma unroll
  for (int t = 0; t < 2; ++t) {
    const int n = (wave * 2 + t) * 16 + m;
    v8f a = {};
    a = wmma_rowtile<IN_CH>(aRowA, W1l, n, khi, a);   // mean_agg @ W1l^T
    a = wmma_rowtile<IN_CH>(aRowX, W1r, n, khi, a);   // + x @ W1r^T
    v8f bb = {};
    bb = wmma_rowtile<IN_CH>(aRowX, Ws, n, khi, bb);  // skip: x @ Ws^T
    const float b1n = b1[n], bsn = bs[n];
#pragma unroll
    for (int r = 0; r < 8; ++r) { a[r] += b1n; bb[r] += bsn; }
    accA[t] = a; accB[t] = bb;
    // partial row sum-of-squares (cross-lane reduce within 16-lane half)
#pragma unroll
    for (int r = 0; r < 8; ++r) {
      float s = a[r] * a[r];
      s += __shfl_xor(s, 1, 16);
      s += __shfl_xor(s, 2, 16);
      s += __shfl_xor(s, 4, 16);
      s += __shfl_xor(s, 8, 16);
      if (m == 0) atomicAdd(&rowsq[r + 8 * khi], s);
    }
  }
  __syncthreads();
  if (tid < 16) rowsq[tid] = 1.0f / fmaxf(sqrtf(rowsq[tid]), 1e-12f);
  __syncthreads();

#pragma unroll
  for (int t = 0; t < 2; ++t) {
    const int n = (wave * 2 + t) * 16 + m;
#pragma unroll
    for (int r = 0; r < 8; ++r) {
      const int mr = r + 8 * khi;
      float v = tanhf(accA[t][r] * rowsq[mr] + accB[t][r]);
      h1[(size_t)(node0 + mr) * HID + n] = v;
    }
  }
}

// ---------------- fused layer 2 + output projection ----------------
// h2 = tanh( normalize(mean_agg2 @ W2l^T + b2 + h1 @ W2r^T) )
// out = h2 @ Wo^T + bo     (h2 kept in LDS as bf16, never spilled to HBM)
__global__ __launch_bounds__(256) void layer2_kernel(
    const float* __restrict__ h1, const float* __restrict__ agg,
    const float* __restrict__ cnt,
    const __bf16* __restrict__ W2l, const float* __restrict__ b2,
    const __bf16* __restrict__ W2r,
    const __bf16* __restrict__ Wo,  const float* __restrict__ bo,
    float* __restrict__ out) {
  __shared__ __align__(16) __bf16 sA[16 * HID];    // mean aggregated h1
  __shared__ __align__(16) __bf16 sX[16 * HID];    // h1 tile
  __shared__ __align__(16) __bf16 sH[16 * OUT_F];  // h2 tile (post tanh)
  __shared__ float rinv[16];
  __shared__ float rowsq[16];

  const int tid = threadIdx.x;
  const int node0 = blockIdx.x * 16;

  if (tid < 16) {
    rinv[tid]  = 1.0f / fmaxf(cnt[node0 + tid], 1.0f);
    rowsq[tid] = 0.0f;
  }
  __syncthreads();

  for (int i = tid; i < 16 * HID; i += 256) {
    int r = i >> 8, k = i & (HID - 1);
    sX[i] = (__bf16)h1[(size_t)(node0 + r) * HID + k];
    sA[i] = (__bf16)(agg[(size_t)(node0 + r) * HID + k] * rinv[r]);
  }
  __syncthreads();

  const int wave = tid >> 5, lane = tid & 31;
  const int m = lane & 15, khi = lane >> 4;
  const __bf16* aRowA = sA + m * HID;
  const __bf16* aRowX = sX + m * HID;

  v8f accA[2];
#pragma unroll
  for (int t = 0; t < 2; ++t) {
    const int n = (wave * 2 + t) * 16 + m;
    v8f a = {};
    a = wmma_rowtile<HID>(aRowA, W2l, n, khi, a);
    a = wmma_rowtile<HID>(aRowX, W2r, n, khi, a);
    const float b2n = b2[n];
#pragma unroll
    for (int r = 0; r < 8; ++r) a[r] += b2n;
    accA[t] = a;
#pragma unroll
    for (int r = 0; r < 8; ++r) {
      float s = a[r] * a[r];
      s += __shfl_xor(s, 1, 16);
      s += __shfl_xor(s, 2, 16);
      s += __shfl_xor(s, 4, 16);
      s += __shfl_xor(s, 8, 16);
      if (m == 0) atomicAdd(&rowsq[r + 8 * khi], s);
    }
  }
  __syncthreads();
  if (tid < 16) rowsq[tid] = 1.0f / fmaxf(sqrtf(rowsq[tid]), 1e-12f);
  __syncthreads();

#pragma unroll
  for (int t = 0; t < 2; ++t) {
    const int n = (wave * 2 + t) * 16 + m;
#pragma unroll
    for (int r = 0; r < 8; ++r) {
      const int mr = r + 8 * khi;
      sH[mr * OUT_F + n] = (__bf16)tanhf(accA[t][r] * rowsq[mr]);
    }
  }
  __syncthreads();

  // Final projection: 16x128 output, 8 n-tiles -> one per wave. K = 256.
  {
    const int n = wave * 16 + m;
    const __bf16* aRowH = sH + m * OUT_F;
    v8f c = {};
    c = wmma_rowtile<OUT_F>(aRowH, Wo, n, khi, c);
    const float bon = bo[n];
#pragma unroll
    for (int r = 0; r < 8; ++r) {
      const int mr = r + 8 * khi;
      out[(size_t)(node0 + mr) * OUT_CH + n] = c[r] + bon;
    }
  }
}

// ---------------- launcher ----------------
extern "C" void kernel_launch(void* const* d_in, const int* in_sizes, int n_in,
                              void* d_out, int out_size, void* d_ws, size_t ws_size,
                              hipStream_t stream) {
  const float* x    = (const float*)d_in[0];
  const int*   ei   = (const int*)d_in[1];      // [2, E]
  const unsigned char* mask = (const unsigned char*)d_in[2];
  const float* W1l  = (const float*)d_in[3];
  const float* b1   = (const float*)d_in[4];
  const float* W1r  = (const float*)d_in[5];
  const float* Ws   = (const float*)d_in[6];
  const float* bs   = (const float*)d_in[7];
  const float* W2l  = (const float*)d_in[8];
  const float* b2   = (const float*)d_in[9];
  const float* W2r  = (const float*)d_in[10];
  const float* Wo   = (const float*)d_in[11];
  const float* bo   = (const float*)d_in[12];
  (void)in_sizes; (void)n_in; (void)out_size; (void)ws_size;

  const int* src = ei;
  const int* dst = ei + N_EDGES;

  // Workspace layout:
  //   floats: agg1 | cnt1 | agg2 | cnt2 | h1
  //   bf16  : W1l | W1r | Ws | W2l | W2r | Wo
  float* ws   = (float*)d_ws;
  float* agg1 = ws;
  float* cnt1 = agg1 + (size_t)N_NODES * IN_CH;
  float* agg2 = cnt1 + N_NODES;
  float* cnt2 = agg2 + (size_t)N_NODES * HID;
  float* h1   = cnt2 + N_NODES;
  __bf16* bw    = (__bf16*)(h1 + (size_t)N_NODES * HID);
  __bf16* W1lb  = bw;                      // 256*128
  __bf16* W1rb  = W1lb + HID * IN_CH;      // 256*128
  __bf16* Wsb   = W1rb + HID * IN_CH;      // 256*128
  __bf16* W2lb  = Wsb  + HID * IN_CH;      // 256*256
  __bf16* W2rb  = W2lb + OUT_F * HID;      // 256*256
  __bf16* Wob   = W2rb + OUT_F * HID;      // 128*256

  // 1) zero segment-sum buffers (agg1..cnt2 contiguous)
  const long zcount = (long)N_NODES * (IN_CH + 1 + HID + 1);
  zero_kernel<<<2048, 256, 0, stream>>>(ws, zcount);

  // 2) pre-convert weights to bf16 (tiny; L2-resident afterwards)
  cvt_bf16_kernel<<<(HID * IN_CH + 255) / 256, 256, 0, stream>>>(W1l, W1lb, HID * IN_CH);
  cvt_bf16_kernel<<<(HID * IN_CH + 255) / 256, 256, 0, stream>>>(W1r, W1rb, HID * IN_CH);
  cvt_bf16_kernel<<<(HID * IN_CH + 255) / 256, 256, 0, stream>>>(Ws, Wsb, HID * IN_CH);
  cvt_bf16_kernel<<<(OUT_F * HID + 255) / 256, 256, 0, stream>>>(W2l, W2lb, OUT_F * HID);
  cvt_bf16_kernel<<<(OUT_F * HID + 255) / 256, 256, 0, stream>>>(W2r, W2rb, OUT_F * HID);
  cvt_bf16_kernel<<<(OUT_CH * OUT_F + 255) / 256, 256, 0, stream>>>(Wo, Wob, OUT_CH * OUT_F);

  // 3) conv1 aggregation (atomic scatter-add, memory/atomic bound)
  {
    long threads = (long)N_EDGES * 32;
    scatter1_kernel<<<(unsigned)((threads + 255) / 256), 256, 0, stream>>>(
        x, src, dst, agg1, cnt1);
  }

  // 4) fused conv1 GEMMs + norm + skip + tanh
  layer1_kernel<<<N_NODES / 16, 256, 0, stream>>>(
      x, agg1, cnt1, W1lb, b1, W1rb, Wsb, bs, h1);

  // 5) conv2 aggregation on edge-dropped graph
  {
    long threads = (long)N_EDGES * 64;
    scatter2_kernel<<<(unsigned)((threads + 255) / 256), 256, 0, stream>>>(
        h1, src, dst, mask, agg2, cnt2);
  }

  // 6) fused conv2 GEMMs + norm + tanh + output projection
  layer2_kernel<<<N_NODES / 16, 256, 0, stream>>>(
      h1, agg2, cnt2, W2lb, b2, W2rb, Wob, bo, (float*)d_out);
}
